// LLES_33638183862957
// MI455X (gfx1250) — matrix operand order repaired and checked by reading, not compiled, not run on hardware
//
#include <hip/hip_runtime.h>
#include <hip/hip_bf16.h>
#include <math.h>

// ---------------------------------------------------------------------------
// Problem constants (from reference)
// ---------------------------------------------------------------------------
#define NPART   262144
#define LNBR    64
#define PI_F    3.14159265358f
#define TWOPI_F 6.28318530716f
// H = ((2*pi)^3 / N * L / pi / (4/3))^(1/3)
#define H_F     0.2436096f
#define INV_H_F (1.0f / H_F)

typedef __attribute__((ext_vector_type(16))) _Float16 v16h;
typedef __attribute__((ext_vector_type(8)))  _Float16 v8h;
typedef __attribute__((ext_vector_type(8)))  float    v8f;

// Weight fragment buffer layout (in d_ws, produced by prep kernel):
//   36 tiles (18 per MLP: vel then rho), each tile = 32 lanes x 16 halves
//   = 512 f16 = 1024 B  -> 36864 B total
//   tile order per MLP: L1 tn0..1 | L2 tn0..6 | L3 (kc,tn) 9+kc*2+tn | L4 (17)
// Biases (f32, zero padded) at byte offset 36864:
//   per MLP 192 floats: b1[32] b2[112] b3[32] b4[16]; vel then rho = 384 floats
#define FRAG_HALVES   (36 * 512)
#define FRAG_BYTES    (FRAG_HALVES * 2)
#define BIAS_FLOATS   384

// ---------------------------------------------------------------------------
// Prep kernel: f32 weights -> f16 B-layout fragments + padded biases
// B fragment element (tile, lane, i):  K = kc*32 + (i&7) + (i>=8)*16 + (lane>=16)*8,
//                                      N = tn*16 + (lane&15)
// ---------------------------------------------------------------------------
__global__ void lles_prep(
    const float* W1, const float* W2, const float* W3, const float* W4,
    const float* Wr1, const float* Wr2, const float* Wr3, const float* Wr4,
    const float* b1, const float* b2, const float* b3, const float* b4,
    const float* br1, const float* br2, const float* br3, const float* br4,
    _Float16* frag, float* bias)
{
    const int idx = blockIdx.x * 256 + threadIdx.x;
    const float* Wsel[8] = {W1, W2, W3, W4, Wr1, Wr2, Wr3, Wr4};
    const float* Bsel[8] = {b1, b2, b3, b4, br1, br2, br3, br4};
    const int fins[4]   = {5, 20, 100, 20};
    const int foutsv[4] = {20, 100, 20, 2};
    const int foutsr[4] = {20, 100, 20, 1};

    if (idx < FRAG_HALVES) {
        const int t    = idx >> 9;        // tile 0..35
        const int e    = idx & 511;
        const int lane = e >> 4;
        const int i    = e & 15;
        const int mlp  = t / 18;
        const int tt   = t % 18;
        int layer, kc, tn;
        if (tt < 2)       { layer = 0; kc = 0;             tn = tt;           }
        else if (tt < 9)  { layer = 1; kc = 0;             tn = tt - 2;       }
        else if (tt < 17) { layer = 2; kc = (tt - 9) >> 1; tn = (tt - 9) & 1; }
        else              { layer = 3; kc = 0;             tn = 0;            }
        const float* W = Wsel[mlp * 4 + layer];
        const int fin  = fins[layer];
        const int fout = mlp ? foutsr[layer] : foutsv[layer];
        const int K = kc * 32 + (i & 7) + ((i >> 3) << 4) + ((lane >= 16) ? 8 : 0);
        const int N = tn * 16 + (lane & 15);
        float v = 0.0f;
        if (K < fin && N < fout) v = W[K * fout + N];
        frag[idx] = (_Float16)v;
    } else if (idx < FRAG_HALVES + BIAS_FLOATS) {
        const int b   = idx - FRAG_HALVES;   // 0..383
        const int mlp = b / 192;
        const int o   = b % 192;
        int layer, off;
        if (o < 32)       { layer = 0; off = o;       }
        else if (o < 144) { layer = 1; off = o - 32;  }
        else if (o < 176) { layer = 2; off = o - 144; }
        else              { layer = 3; off = o - 176; }
        const int fout = mlp ? foutsr[layer] : foutsv[layer];
        bias[b] = (off < fout) ? Bsel[mlp * 4 + layer][off] : 0.0f;
    }
}

// ---------------------------------------------------------------------------
// Device helpers
// ---------------------------------------------------------------------------
__device__ __forceinline__ void lds_fence() {
    // Wave-local LDS ordering: DS ops are in-order per wave; this stops the
    // compiler from reordering and drains DScnt (CDNA5 split counter).
    __asm__ volatile("s_wait_dscnt 0" ::: "memory");
}

// CDNA5 has a native V_TANH_F32 transcendental (TRANS32, co-executes with the
// matrix pipe). Use it when the toolchain exposes the builtin; otherwise use a
// branch-free exp2/rcp formulation (no EXEC-mask branching like OCML tanhf).
__device__ __forceinline__ float fast_tanh(float x) {
#if __has_builtin(__builtin_amdgcn_tanhf)
    return __builtin_amdgcn_tanhf(x);
#elif __has_builtin(__builtin_amdgcn_tanh_f32)
    return __builtin_amdgcn_tanh_f32(x);
#else
    // tanh(x) = sign(x) * (1 - t) / (1 + t),  t = exp(-2|x|)
#if __has_builtin(__builtin_amdgcn_exp2f)
    const float t = __builtin_amdgcn_exp2f(-2.885390082f * fabsf(x));
#else
    const float t = __expf(-2.0f * fabsf(x));
#endif
#if __has_builtin(__builtin_amdgcn_rcpf)
    const float r = (1.0f - t) * __builtin_amdgcn_rcpf(1.0f + t);
#else
    const float r = (1.0f - t) / (1.0f + t);
#endif
    return copysignf(r, x);
#endif
}

__device__ __forceinline__ v8f wmma16(v16h a, v16h b, v8f c) {
    return __builtin_amdgcn_wmma_f32_16x16x32_f16(
        false, a, false, b, (short)0, c, false, false);
}

__device__ __forceinline__ v16h load_bfrag(const _Float16* wf, int tile, int lane) {
    return *reinterpret_cast<const v16h*>(wf + tile * 512 + lane * 16);
}

// A fragment (16x32 f16) from a 16x128 LDS activation matrix at K-chunk kc.
// lane<16: a[0..7]=K0..7, a[8..15]=K16..23 ; lane>=16: K base +8.
__device__ __forceinline__ v16h load_afrag(const _Float16* hb, int lane, int kc) {
    const int M    = lane & 15;
    const int kadd = (lane >> 4) << 3;
    const _Float16* row = hb + M * 128 + kc * 32 + kadd;
    v8h lo = *reinterpret_cast<const v8h*>(row);
    v8h hi = *reinterpret_cast<const v8h*>(row + 16);
    v16h a;
#pragma unroll
    for (int i = 0; i < 8; ++i) { a[i] = lo[i]; a[i + 8] = hi[i]; }
    return a;
}

// C layout: lane holds column N=(lane&15) of tile tn, rows (lane>=16)*8 + r.
__device__ __forceinline__ void store_tanh(_Float16* hb, int tn, int lane,
                                           v8f c, float bias) {
    const int N  = tn * 16 + (lane & 15);
    const int Mb = (lane >> 4) << 3;
#pragma unroll
    for (int r = 0; r < 8; ++r)
        hb[(Mb + r) * 128 + N] = (_Float16)fast_tanh(c[r] + bias);
}

// ---------------------------------------------------------------------------
// Dual-column MLP: runs one MLP (5->20->100->20->ncols) on TWO 16-row feature
// fragments, sharing every B-fragment LDS load between the two columns
// (halves the weight-reload LDS bandwidth, which otherwise outpaces the
// matrix pipe: 36 KB/column vs ~64 B/cycle/SIMD of LDS read bandwidth).
// ---------------------------------------------------------------------------
__device__ void run_mlp2(const _Float16* wf, const float* wb, int tbase, int bbase,
                         v16h a0A, v16h a0B, _Float16* hbA, _Float16* hbB,
                         float* stg, int lane, int ncols,
                         float* koutA, float* koutB)
{
    const int ncol = lane & 15;
    // ---- layer 1: [16,32(K pad 5)] x [32,32(N pad 20)] ----
#pragma unroll
    for (int tn = 0; tn < 2; ++tn) {
        v16h b = load_bfrag(wf, tbase + tn, lane);
        v8f cA = {}; cA = wmma16(a0A, b, cA);
        v8f cB = {}; cB = wmma16(a0B, b, cB);
        const float bias = wb[bbase + tn * 16 + ncol];
        store_tanh(hbA, tn, lane, cA, bias);
        store_tanh(hbB, tn, lane, cB, bias);
    }
    lds_fence();
    v16h a1A = load_afrag(hbA, lane, 0);
    v16h a1B = load_afrag(hbB, lane, 0);
    lds_fence();
    // ---- layer 2: [16,32(K pad 20)] x [32,112(N pad 100)] ----
#pragma unroll
    for (int tn = 0; tn < 7; ++tn) {
        v16h b = load_bfrag(wf, tbase + 2 + tn, lane);
        v8f cA = {}; cA = wmma16(a1A, b, cA);
        v8f cB = {}; cB = wmma16(a1B, b, cB);
        const float bias = wb[bbase + 32 + tn * 16 + ncol];
        store_tanh(hbA, tn, lane, cA, bias);
        store_tanh(hbB, tn, lane, cB, bias);
    }
    lds_fence();
    v16h a2A0 = load_afrag(hbA, lane, 0);
    v16h a2A1 = load_afrag(hbA, lane, 1);
    v16h a2A2 = load_afrag(hbA, lane, 2);
    v16h a2A3 = load_afrag(hbA, lane, 3);
    v16h a2B0 = load_afrag(hbB, lane, 0);
    v16h a2B1 = load_afrag(hbB, lane, 1);
    v16h a2B2 = load_afrag(hbB, lane, 2);
    v16h a2B3 = load_afrag(hbB, lane, 3);
    lds_fence();
    // ---- layer 3: [16,128(K pad 100)] x [128,32(N pad 20)] ----
#pragma unroll
    for (int tn = 0; tn < 2; ++tn) {
        v8f cA = {}, cB = {};
        {
            v16h b = load_bfrag(wf, tbase + 9 + 0 * 2 + tn, lane);
            cA = wmma16(a2A0, b, cA); cB = wmma16(a2B0, b, cB);
        }
        {
            v16h b = load_bfrag(wf, tbase + 9 + 1 * 2 + tn, lane);
            cA = wmma16(a2A1, b, cA); cB = wmma16(a2B1, b, cB);
        }
        {
            v16h b = load_bfrag(wf, tbase + 9 + 2 * 2 + tn, lane);
            cA = wmma16(a2A2, b, cA); cB = wmma16(a2B2, b, cB);
        }
        {
            v16h b = load_bfrag(wf, tbase + 9 + 3 * 2 + tn, lane);
            cA = wmma16(a2A3, b, cA); cB = wmma16(a2B3, b, cB);
        }
        const float bias = wb[bbase + 144 + tn * 16 + ncol];
        store_tanh(hbA, tn, lane, cA, bias);
        store_tanh(hbB, tn, lane, cB, bias);
    }
    lds_fence();
    v16h a3A = load_afrag(hbA, lane, 0);
    v16h a3B = load_afrag(hbB, lane, 0);
    lds_fence();
    // ---- layer 4: [16,32(K pad 20)] x [32,16(N pad ncols)] ----
    {
        v16h b = load_bfrag(wf, tbase + 17, lane);
        v8f cA = {}; cA = wmma16(a3A, b, cA);
        v8f cB = {}; cB = wmma16(a3B, b, cB);
        const int Mb = (lane >> 4) << 3;
        if (ncol < ncols) {
            const float bias = wb[bbase + 176 + ncol];
#pragma unroll
            for (int r = 0; r < 8; ++r) {
                stg[(Mb + r) * 4 + ncol]     = cA[r] + bias;
                stg[(Mb + r) * 4 + 2 + ncol] = cB[r] + bias;
            }
        }
    }
    lds_fence();
    koutA[0] = 0.0f; koutA[1] = 0.0f; koutB[0] = 0.0f; koutB[1] = 0.0f;
    if (lane < 16) {
        koutA[0] = stg[lane * 4 + 0];
        koutB[0] = stg[lane * 4 + 2];
        if (ncols > 1) {
            koutA[1] = stg[lane * 4 + 1];
            koutB[1] = stg[lane * 4 + 3];
        }
    }
}

__device__ __forceinline__ float sgnf(float x) {
    return (x > 0.0f) ? 1.0f : ((x < 0.0f) ? -1.0f : 0.0f);
}

// Periodic minimum-image displacement, scaled by 1/H (cal_disv).
__device__ __forceinline__ float pdisp(float dx) {
    const float t1 = fabsf(dx);
    const float s  = -sgnf(dx) * sgnf(dx + PI_F) * sgnf(dx - PI_F);
    return s * fminf(t1, TWOPI_F - t1) * INV_H_F;
}

struct Geom {
    float dsign, dx0, dx1, dx2, dv0, dv1, dv2, orho, oav;
};

// Per-pair geometry + feature fragment (lanes 0..15 only; rest zero).
__device__ __forceinline__ v16h make_feat(
    const float* __restrict__ X, const float* __restrict__ V,
    const float* __restrict__ rho, const int* __restrict__ neighbor,
    int p, int j, bool act,
    float Xb0, float Xb1, float Xb2, float Vb0, float Vb1, float Vb2, float rb,
    float A1, float A2, float B1, float B2, Geom& g)
{
    v16h a0 = {};
    g.dsign = 0; g.dx0 = 0; g.dx1 = 0; g.dx2 = 0;
    g.dv0 = 0; g.dv1 = 0; g.dv2 = 0; g.orho = 0; g.oav = 0;
    if (act) {
        const float h2 = H_F * H_F;
        const int nbr = neighbor[p * LNBR + j];
        const float o0 = pdisp(Xb0 - X[nbr * 3 + 0]);
        const float o1 = pdisp(Xb1 - X[nbr * 3 + 1]);
        const float o2 = pdisp(Xb2 - X[nbr * 3 + 2]);
        const float w0 = Vb0 - V[nbr * 3 + 0];
        const float w1 = Vb1 - V[nbr * 3 + 1];
        const float w2 = Vb2 - V[nbr * 3 + 2];
        const float d2 = rho[nbr];
        const float out2  = o0 * o0 + o1 * o1 + o2 * o2;
        const float outv2 = w0 * w0 + w1 * w1 + w2 * w2;
        const float out2v = o0 * w0 + o1 * w1 + o2 * w2;
        const float r  = sqrtf(out2);
        const float rv = sqrtf(outv2);
        const float dd = rb - d2;
        g.dsign = dd / fabsf(dd);
        const float ir = 1.0f / r, irv = 1.0f / rv;
        g.dx0 = o0 * ir;  g.dx1 = o1 * ir;  g.dx2 = o2 * ir;
        g.dv0 = w0 * irv; g.dv1 = w1 * irv; g.dv2 = w2 * irv;
        const float den = out2 + 0.1f * h2;
        const float t = dd * h2 / den;
        g.orho = -(B1 + B2 * fabsf(t)) * t;
        const float o = -H_F * fast_tanh(-out2v) / den;
        g.oav = -A1 * o + A2 * o * o;
        a0[0] = (_Float16)rb;  a0[1] = (_Float16)d2;
        a0[2] = (_Float16)r;   a0[3] = (_Float16)rv;
        a0[4] = (_Float16)out2v;
    }
    return a0;
}

// ---------------------------------------------------------------------------
// Main kernel: 256 threads = 8 wave32; one wave = 16 particles; 32 dual-column
// iterations cover neighbor columns 1..63 (last pair has its 2nd column
// masked). 72 WMMAs per iteration per wave (2 columns x (18 vel + 18 rho)).
// ---------------------------------------------------------------------------
__global__ __launch_bounds__(256) void lles_main(
    const float* __restrict__ X, const float* __restrict__ V,
    const float* __restrict__ rho, const int* __restrict__ neighbor,
    const float* __restrict__ alpha1, const float* __restrict__ alpha2,
    const float* __restrict__ beta1, const float* __restrict__ beta2,
    const _Float16* __restrict__ frag_g, const float* __restrict__ bias_g,
    float* __restrict__ out)
{
    __shared__ alignas(32) _Float16 wfrag[FRAG_HALVES];   // 36 KB weight frags
    __shared__ float wbias[BIAS_FLOATS];                  // 1.5 KB biases
    __shared__ alignas(32) _Float16 hbuf[8][2][2048];     // 64 KB activations
    __shared__ float stage[8][64];                        // 2 KB output stage

    const int tid  = threadIdx.x;
    const int wave = tid >> 5;
    const int lane = tid & 31;

    // Stage weights into LDS (128-bit DS stores).
    {
        const int4* src = reinterpret_cast<const int4*>(frag_g);
        int4* dst = reinterpret_cast<int4*>(wfrag);
        for (int i = tid; i < FRAG_BYTES / 16; i += 256) dst[i] = src[i];
        for (int i = tid; i < BIAS_FLOATS; i += 256) wbias[i] = bias_g[i];
    }
    // Zero this wave's activation buffers (cols 112..127 must stay 0 forever).
    {
        _Float16* hz = &hbuf[wave][0][0];
        for (int i = lane; i < 4096; i += 32) hz[i] = (_Float16)0.0f;
    }
    __syncthreads();

    _Float16* hbA = hbuf[wave][0];
    _Float16* hbB = hbuf[wave][1];
    float* stg = stage[wave];
    const bool act = (lane < 16);
    const int p = blockIdx.x * 128 + wave * 16 + (lane & 15);

    float Xb0 = 0, Xb1 = 0, Xb2 = 0, Vb0 = 0, Vb1 = 0, Vb2 = 0, rb = 0;
    if (act) {
        Xb0 = X[p * 3 + 0]; Xb1 = X[p * 3 + 1]; Xb2 = X[p * 3 + 2];
        Vb0 = V[p * 3 + 0]; Vb1 = V[p * 3 + 1]; Vb2 = V[p * 3 + 2];
        rb  = rho[p];
    }
    const float A1 = fabsf(alpha1[0]), A2 = fabsf(alpha2[0]);
    const float B1 = fabsf(beta1[0]),  B2 = fabsf(beta2[0]);

    float acc0 = 0, accx = 0, accy = 0, accz = 0;

#pragma unroll 1
    for (int ji = 0; ji < 32; ++ji) {
        const int  jA = 1 + 2 * ji;
        const bool validB = (2 + 2 * ji) < LNBR;
        const int  jB = validB ? (2 + 2 * ji) : (LNBR - 1);

        Geom gA, gB;
        v16h a0A = make_feat(X, V, rho, neighbor, p, jA, act,
                             Xb0, Xb1, Xb2, Vb0, Vb1, Vb2, rb, A1, A2, B1, B2, gA);
        v16h a0B = make_feat(X, V, rho, neighbor, p, jB, act,
                             Xb0, Xb1, Xb2, Vb0, Vb1, Vb2, rb, A1, A2, B1, B2, gB);

        float kvA[2], kvB[2], krA[2], krB[2];
        run_mlp2(wfrag, wbias, 0, 0, a0A, a0B, hbA, hbB, stg, lane, 2, kvA, kvB);
        run_mlp2(wfrag, wbias, 18, 192, a0A, a0B, hbA, hbB, stg, lane, 1, krA, krB);

        if (act) {
            acc0 += krA[0] * gA.dsign + gA.orho;
            accx += kvA[0] * gA.dx0 + kvA[1] * gA.dv0 + gA.oav * gA.dx0;
            accy += kvA[0] * gA.dx1 + kvA[1] * gA.dv1 + gA.oav * gA.dx1;
            accz += kvA[0] * gA.dx2 + kvA[1] * gA.dv2 + gA.oav * gA.dx2;
            if (validB) {
                acc0 += krB[0] * gB.dsign + gB.orho;
                accx += kvB[0] * gB.dx0 + kvB[1] * gB.dv0 + gB.oav * gB.dx0;
                accy += kvB[0] * gB.dx1 + kvB[1] * gB.dv1 + gB.oav * gB.dx1;
                accz += kvB[0] * gB.dx2 + kvB[1] * gB.dv2 + gB.oav * gB.dx2;
            }
        }
    }
    if (act) {
        out[p * 4 + 0] = acc0;
        out[p * 4 + 1] = accx;
        out[p * 4 + 2] = accy;
        out[p * 4 + 3] = accz;
    }
}

// ---------------------------------------------------------------------------
// Host entry
// ---------------------------------------------------------------------------
extern "C" void kernel_launch(void* const* d_in, const int* in_sizes, int n_in,
                              void* d_out, int out_size, void* d_ws, size_t ws_size,
                              hipStream_t stream)
{
    (void)in_sizes; (void)n_in; (void)out_size; (void)ws_size;
    const float* X        = (const float*)d_in[0];
    const float* V        = (const float*)d_in[1];
    const float* rho      = (const float*)d_in[2];
    const float* W1  = (const float*)d_in[3];  const float* b1  = (const float*)d_in[4];
    const float* W2  = (const float*)d_in[5];  const float* b2  = (const float*)d_in[6];
    const float* W3  = (const float*)d_in[7];  const float* b3  = (const float*)d_in[8];
    const float* W4  = (const float*)d_in[9];  const float* b4  = (const float*)d_in[10];
    const float* Wr1 = (const float*)d_in[11]; const float* br1 = (const float*)d_in[12];
    const float* Wr2 = (const float*)d_in[13]; const float* br2 = (const float*)d_in[14];
    const float* Wr3 = (const float*)d_in[15]; const float* br3 = (const float*)d_in[16];
    const float* Wr4 = (const float*)d_in[17]; const float* br4 = (const float*)d_in[18];
    const float* alpha1 = (const float*)d_in[19];
    const float* alpha2 = (const float*)d_in[20];
    const float* beta1  = (const float*)d_in[21];
    const float* beta2  = (const float*)d_in[22];
    const int* neighbor = (const int*)d_in[23];

    _Float16* frag = (_Float16*)d_ws;
    float* biasb = (float*)((char*)d_ws + FRAG_BYTES);

    const int prep_work = FRAG_HALVES + BIAS_FLOATS;
    lles_prep<<<(prep_work + 255) / 256, 256, 0, stream>>>(
        W1, W2, W3, W4, Wr1, Wr2, Wr3, Wr4,
        b1, b2, b3, b4, br1, br2, br3, br4, frag, biasb);

    lles_main<<<NPART / 128, 256, 0, stream>>>(
        X, V, rho, neighbor, alpha1, alpha2, beta1, beta2,
        frag, biasb, (float*)d_out);
}
